// OffloadedNemotronMoE_48335561949264
// MI455X (gfx1250) — matrix operand order repaired and checked by reading, not compiled, not run on hardware
//
#include <hip/hip_runtime.h>
#include <math.h>

// ---------------------------------------------------------------------------
// MoE (Nemotron-style): router top-2 + dense per-expert SwiGLU FFN + shared
// expert, computed with v_wmma_f32_16x16x32_bf16 on gfx1250 (wave32).
//
// T=512, H=1024, I=1024, E=8, top-2. Weights (~110MB fp32) stream once and
// are converted to bf16 while staging into LDS; inter-GEMM activations are
// bf16 in workspace. Register-staged prefetch overlaps HBM latency with the
// WMMA burst of the current K-step.
// ---------------------------------------------------------------------------

#define T_ 512
#define H_ 1024
#define I_ 1024
#define E_ 8

#define BM   128   // token tile
#define BKA  32    // K step (bf16 WMMA K=32)
#define BN1  64    // GEMM1 output tile (gate & up computed together)
#define BN2  128   // GEMM2 output tile
#define LDSK 40    // LDS row stride in ushorts (80B: 16B-aligned chunks, spreads banks)

typedef __attribute__((ext_vector_type(16))) __bf16        v16bf;
typedef __attribute__((ext_vector_type(8)))  float         v8f;
typedef __attribute__((ext_vector_type(4)))  float         f32x4;
typedef __attribute__((ext_vector_type(4)))  unsigned int  u32x4;

__device__ __forceinline__ unsigned short f2bf(float f) {
    union { float f; unsigned int u; } c; c.f = f;
    unsigned int u = c.u;
    // round-to-nearest-even fp32 -> bf16
    return (unsigned short)((u + 0x7FFFu + ((u >> 16) & 1u)) >> 16);
}

// Pack two fp32 into packed bf16x2; prefer the single-instruction packed
// convert if this toolchain exposes it for gfx1250.
#if defined(__has_builtin)
#if __has_builtin(__builtin_amdgcn_cvt_pk_bf16_f32)
#define HAVE_CVT_PK_BF16 1
#endif
#endif

__device__ __forceinline__ unsigned int pack2bf(float lo, float hi) {
#ifdef HAVE_CVT_PK_BF16
    auto r = __builtin_amdgcn_cvt_pk_bf16_f32(lo, hi);
    return __builtin_bit_cast(unsigned int, r);
#else
    return (unsigned int)f2bf(lo) | ((unsigned int)f2bf(hi) << 16);
#endif
}

// ---- register staging helpers ---------------------------------------------
__device__ __forceinline__ void ld16f(f32x4 r[4], const float* src) {
    const f32x4* s = (const f32x4*)src;
    r[0] = s[0]; r[1] = s[1]; r[2] = s[2]; r[3] = s[3];
}
__device__ __forceinline__ void st16bf(unsigned short* dst, const f32x4 r[4]) {
    u32x4 q0, q1;
    q0.x = pack2bf(r[0].x, r[0].y); q0.y = pack2bf(r[0].z, r[0].w);
    q0.z = pack2bf(r[1].x, r[1].y); q0.w = pack2bf(r[1].z, r[1].w);
    q1.x = pack2bf(r[2].x, r[2].y); q1.y = pack2bf(r[2].z, r[2].w);
    q1.z = pack2bf(r[3].x, r[3].y); q1.w = pack2bf(r[3].z, r[3].w);
    ((u32x4*)dst)[0] = q0;
    ((u32x4*)dst)[1] = q1;
}
__device__ __forceinline__ void ld32B(u32x4 r[2], const unsigned short* src) {
    r[0] = ((const u32x4*)src)[0];
    r[1] = ((const u32x4*)src)[1];
}
__device__ __forceinline__ void st32B(unsigned short* dst, const u32x4 r[2]) {
    ((u32x4*)dst)[0] = r[0];
    ((u32x4*)dst)[1] = r[1];
}

// ---- WMMA fragment loads (16-bit A/B lane layouts per CDNA5 ISA 7.12.2) ----
// A fragment (16x32, M x K): lane l holds row M=l%16.
// Lanes 0-15: K {0..7, 16..23}; lanes 16-31: K {8..15, 24..31}.
__device__ __forceinline__ v16bf load_frag_a(const unsigned short* lds, int row0, int lane) {
    const unsigned short* p = lds + (row0 + (lane & 15)) * LDSK;
    const int hi = (lane >> 4) << 3;  // 0 or 8
    union { v16bf v; u32x4 q[2]; } u;
    u.q[0] = *(const u32x4*)(p + hi);
    u.q[1] = *(const u32x4*)(p + 16 + hi);
    return u.v;
}
// B fragment (32x16, K x N), LDS stored [n][k]: lane l holds column N=l%16,
// Lanes 0-15: K {0..7, 16..23}; lanes 16-31: K {8..15, 24..31}.
__device__ __forceinline__ v16bf load_frag_b(const unsigned short* lds, int n0, int lane) {
    const unsigned short* p = lds + (n0 + (lane & 15)) * LDSK + ((lane >> 4) << 3);
    union { v16bf v; u32x4 q[2]; } u;
    u.q[0] = *(const u32x4*)(p);
    u.q[1] = *(const u32x4*)(p + 16);
    return u.v;
}

__device__ __forceinline__ v8f wmma_bf16(v16bf a, v16bf b, v8f c) {
    return __builtin_amdgcn_wmma_f32_16x16x32_bf16(
        /*neg_a=*/false, a, /*neg_b=*/false, b,
        /*c_mod=*/(short)0, c, /*reuse_a=*/false, /*reuse_b=*/false);
}

// ---------------------------------------------------------------------------
// Router: logits = x @ gate_w^T + bias; top-2, renormalized (softmax cancels).
// ---------------------------------------------------------------------------
__global__ void router_topk(const float* __restrict__ x, const float* __restrict__ gw,
                            const float* __restrict__ bias, float* __restrict__ combine) {
    int t = blockIdx.x * blockDim.x + threadIdx.x;
    if (t >= T_) return;
    float lg[E_];
    const f32x4* xp = (const f32x4*)(x + (size_t)t * H_);
    for (int e = 0; e < E_; ++e) {
        const f32x4* wp = (const f32x4*)(gw + (size_t)e * H_);
        float s = 0.f;
        for (int j = 0; j < H_ / 4; ++j) {
            f32x4 a = xp[j], b = wp[j];
            s += a.x * b.x + a.y * b.y + a.z * b.z + a.w * b.w;
        }
        lg[e] = s + bias[e];
    }
    int i1 = 0;
    for (int e = 1; e < E_; ++e) if (lg[e] > lg[i1]) i1 = e;
    int i2 = (i1 == 0) ? 1 : 0;
    for (int e = 0; e < E_; ++e) if (e != i1 && lg[e] > lg[i2]) i2 = e;
    float wa = 1.0f;
    float wb = expf(lg[i2] - lg[i1]);
    float s  = wa + wb;
    for (int e = 0; e < E_; ++e) combine[t * E_ + e] = 0.f;
    combine[t * E_ + i1] = wa / s;
    combine[t * E_ + i2] = wb / s;
}

// ---------------------------------------------------------------------------
// GEMM1 + SwiGLU. grid = (I/BN1, T/BM, E+1); z==E_ is the shared expert.
// Block tile 128x64; 8 waves; each wave owns 64x16 and computes gate (rows i)
// and up (rows I+i) together, sharing A fragments. Software pipeline: store
// staged tile -> barrier -> issue next tile's loads -> WMMA -> barrier.
// ---------------------------------------------------------------------------
__global__ __launch_bounds__(256) void moe_gemm1_swiglu(
    const float* __restrict__ x, const float* __restrict__ w13,
    const float* __restrict__ shared_w13, const float* __restrict__ combine,
    unsigned short* __restrict__ actw) {
    __shared__ unsigned short lA [BM  * LDSK];
    __shared__ unsigned short lBg[BN1 * LDSK];
    __shared__ unsigned short lBu[BN1 * LDSK];

    const int e     = blockIdx.z;
    const int tBase = blockIdx.y * BM;
    const int iBase = blockIdx.x * BN1;
    const float* W  = (e < E_) ? (w13 + (size_t)e * (2 * I_) * H_) : shared_w13;

    const int tid  = threadIdx.x;
    const int lane = tid & 31;
    const int wave = tid >> 5;
    const int m0   = (wave & 1) * 64;   // 2 wave rows of 64 tokens
    const int n0   = (wave >> 1) * 16;  // 4 wave cols of 16 outputs

    v8f cg[4] = {};
    v8f cu[4] = {};

    const int fr = tid >> 1,   fh = (tid & 1) * 16;   // A fill: 128 rows x 2 halves
    const int tb = tid & 127;
    const int br = tb >> 1,    bh = (tb & 1) * 16;    // B fill: 64 rows x 2 halves

    const float*    pA = x + (size_t)(tBase + fr) * H_ + fh;
    const float*    pB = (tid < 128) ? (W + (size_t)(iBase + br) * H_ + bh)
                                     : (W + (size_t)(I_ + iBase + br) * H_ + bh);
    unsigned short* dA = &lA[fr * LDSK + fh];
    unsigned short* dB = (tid < 128) ? &lBg[br * LDSK + bh] : &lBu[br * LDSK + bh];

    f32x4 rA[4], rB[4];
    ld16f(rA, pA);
    ld16f(rB, pB);

    for (int k0 = 0; k0 < H_; k0 += BKA) {
        st16bf(dA, rA);
        st16bf(dB, rB);
        __syncthreads();
        if (k0 + BKA < H_) {                 // issue next tile's loads early
            ld16f(rA, pA + k0 + BKA);
            ld16f(rB, pB + k0 + BKA);
        }
        v16bf bg = load_frag_b(lBg, n0, lane);
        v16bf bu = load_frag_b(lBu, n0, lane);
        #pragma unroll
        for (int mf = 0; mf < 4; ++mf) {
            v16bf a = load_frag_a(lA, m0 + mf * 16, lane);
            cg[mf] = wmma_bf16(a, bg, cg[mf]);
            cu[mf] = wmma_bf16(a, bu, cu[mf]);
        }
        __syncthreads();                     // LDS reads done before next store
    }

    const int rb  = 8 * (lane >> 4);
    const int col = iBase + n0 + (lane & 15);
    #pragma unroll
    for (int mf = 0; mf < 4; ++mf) {
        #pragma unroll
        for (int v = 0; v < 8; ++v) {
            int   row = tBase + m0 + mf * 16 + rb + v;
            float sc  = (e < E_) ? combine[row * E_ + e] : 1.0f;
            float g   = cg[mf][v];
            float u   = cu[mf][v];
            float act = (g / (1.0f + expf(-g))) * u * sc;
            actw[((size_t)e * T_ + row) * I_ + col] = f2bf(act);
        }
    }
}

// ---------------------------------------------------------------------------
// GEMM2 + combine: out[t,h] = sum_{e=0..8} actw[e] @ W2e^T (scales already in
// actw). grid = (H/BN2, T/BM); block tile 128x128, wave 64x32. The expert and
// K loops are flattened into one 288-step pipeline so prefetch never drains.
// ---------------------------------------------------------------------------
__global__ __launch_bounds__(256) void moe_gemm2_combine(
    const unsigned short* __restrict__ actw, const float* __restrict__ w2,
    const float* __restrict__ shared_w2, float* __restrict__ out) {
    __shared__ unsigned short lA[BM  * LDSK];
    __shared__ unsigned short lB[BN2 * LDSK];

    const int tBase = blockIdx.y * BM;
    const int hBase = blockIdx.x * BN2;
    const int tid  = threadIdx.x;
    const int lane = tid & 31;
    const int wave = tid >> 5;
    const int m0   = (wave & 1) * 64;
    const int n0   = (wave >> 1) * 32;

    v8f c[4][2] = {};

    const int fr = tid >> 1, fh = (tid & 1) * 16;
    constexpr int KSTEPS = I_ / BKA;          // 32
    constexpr int NSTEP  = (E_ + 1) * KSTEPS; // 288

    unsigned short* dA = &lA[fr * LDSK + fh];
    unsigned short* dB = &lB[fr * LDSK + fh];

    auto aPtr = [&](int step) -> const unsigned short* {
        int ee = step >> 5, k0 = (step & 31) * BKA;
        return actw + (size_t)ee * T_ * I_ + (size_t)(tBase + fr) * I_ + k0 + fh;
    };
    auto bPtr = [&](int step) -> const float* {
        int ee = step >> 5, k0 = (step & 31) * BKA;
        const float* W = (ee < E_) ? (w2 + (size_t)ee * H_ * I_) : shared_w2;
        return W + (size_t)(hBase + fr) * I_ + k0 + fh;
    };

    u32x4 rA[2];
    f32x4 rB[4];
    ld32B(rA, aPtr(0));
    ld16f(rB, bPtr(0));

    for (int step = 0; step < NSTEP; ++step) {
        st32B(dA, rA);
        st16bf(dB, rB);
        __syncthreads();
        if (step + 1 < NSTEP) {              // issue next tile's loads early
            ld32B(rA, aPtr(step + 1));
            ld16f(rB, bPtr(step + 1));
        }
        v16bf b0 = load_frag_b(lB, n0,      lane);
        v16bf b1 = load_frag_b(lB, n0 + 16, lane);
        #pragma unroll
        for (int mf = 0; mf < 4; ++mf) {
            v16bf a = load_frag_a(lA, m0 + mf * 16, lane);
            c[mf][0] = wmma_bf16(a, b0, c[mf][0]);
            c[mf][1] = wmma_bf16(a, b1, c[mf][1]);
        }
        __syncthreads();
    }

    const int rb = 8 * (lane >> 4);
    #pragma unroll
    for (int mf = 0; mf < 4; ++mf) {
        #pragma unroll
        for (int nf = 0; nf < 2; ++nf) {
            int col = hBase + n0 + nf * 16 + (lane & 15);
            #pragma unroll
            for (int v = 0; v < 8; ++v) {
                int row = tBase + m0 + mf * 16 + rb + v;
                out[(size_t)row * H_ + col] = c[mf][nf][v];
            }
        }
    }
}

// ---------------------------------------------------------------------------
extern "C" void kernel_launch(void* const* d_in, const int* in_sizes, int n_in,
                              void* d_out, int out_size, void* d_ws, size_t ws_size,
                              hipStream_t stream) {
    const float* x          = (const float*)d_in[0];  // [T,H]
    const float* gate_w     = (const float*)d_in[1];  // [E,H]
    const float* score_bias = (const float*)d_in[2];  // [E]
    const float* w13        = (const float*)d_in[3];  // [E,2I,H]
    const float* w2         = (const float*)d_in[4];  // [E,H,I]
    const float* shared_w13 = (const float*)d_in[5];  // [2I,H]
    const float* shared_w2  = (const float*)d_in[6];  // [H,I]
    float*       out        = (float*)d_out;          // [T,H]

    // Workspace: combine [T,E] fp32 (16KB), then actw [(E+1),T,I] bf16 (~9.4MB).
    float*          combine = (float*)d_ws;
    unsigned short* actw    = (unsigned short*)((char*)d_ws + (size_t)T_ * E_ * sizeof(float));

    router_topk<<<dim3((T_ + 255) / 256), dim3(256), 0, stream>>>(x, gate_w, score_bias, combine);

    dim3 g1(I_ / BN1, T_ / BM, E_ + 1);  // (16, 4, 9)
    moe_gemm1_swiglu<<<g1, dim3(256), 0, stream>>>(x, w13, shared_w13, combine, actw);

    dim3 g2(H_ / BN2, T_ / BM);          // (8, 4)
    moe_gemm2_combine<<<g2, dim3(256), 0, stream>>>(actw, w2, shared_w2, out);
}
